// DSGM_82411832475898
// MI455X (gfx1250) — matrix operand order repaired and checked by reading, not compiled, not run on hardware
//
#include <hip/hip_runtime.h>
#include <math.h>

typedef __attribute__((ext_vector_type(16))) _Float16 v16h;
typedef __attribute__((ext_vector_type(8)))  float    v8f;
typedef __attribute__((ext_vector_type(4)))  unsigned u32x4;
typedef __attribute__((ext_vector_type(2)))  unsigned u32x2;
typedef __attribute__((ext_vector_type(4)))  _Float16 h4;

constexpr int B_ = 8192, X_ = 512, H_ = 256, Z_ = 64, L_ = 16, W_ = 32;
constexpr float STD_F = 1.0025031276057951f;   // exp(0.0025)

// ---------------------------------------------------------------------------
// Generic WMMA GEMM:  C[M,N] = act( A[M,K] @ W[N,K]^T + bias[N] )
// Software-pipelined with double-buffered LDS: global loads for K-slice k+1
// are issued before the WMMAs for slice k, converted/stored after.
// Block tile 128x64, 8 wave32 waves, wave tile 32x32 (2x2 WMMA).
// act: 0=none, 1=relu, 2=sigmoid.  Requires K%32==0, N%32==0, M%128==0.
// ---------------------------------------------------------------------------
__launch_bounds__(256)
__global__ void gemm_wmma(const float* __restrict__ A, int lda,
                          const float* __restrict__ Wt, int ldw,
                          const float* __restrict__ bias,
                          float* __restrict__ C, int ldc,
                          int M, int N, int K, int act)
{
    __shared__ __align__(16) _Float16 As [2][128 * 40];   // 80B row pitch
    __shared__ __align__(16) _Float16 Bs2[2][32 * 72];    // transposed (k,n), 144B pitch

    const int tid  = threadIdx.x;
    const int lane = tid & 31;
    const int wv   = tid >> 5;
    const int wm   = wv >> 1;          // 0..3
    const int wn   = wv & 1;           // 0..1
    const int m0   = blockIdx.x * 128;
    const int n0   = blockIdx.y * 64;

    // staging assignments
    const int r4 = tid >> 3;           // A row 0..31 (+32*it)
    const int c4 = (tid & 7) * 4;      // A col group
    const int kk = tid & 31;           // W k index
    const int ng = (tid >> 5) * 4;     // W n group (+32*it)

    float4 fa[4];
    float  fw[2][4];

    auto load_slice = [&](int k0) {
#pragma unroll
        for (int it = 0; it < 4; ++it)
            fa[it] = *reinterpret_cast<const float4*>(
                A + (size_t)(m0 + r4 + it * 32) * lda + k0 + c4);
#pragma unroll
        for (int it = 0; it < 2; ++it) {
            if (n0 + it * 32 < N) {            // uniform per half-tile (N%32==0)
                const float* wr = Wt + (size_t)(n0 + ng + it * 32) * ldw + k0 + kk;
                fw[it][0] = wr[0];
                fw[it][1] = wr[(size_t)ldw];
                fw[it][2] = wr[(size_t)ldw * 2];
                fw[it][3] = wr[(size_t)ldw * 3];
            } else {
                fw[it][0] = fw[it][1] = fw[it][2] = fw[it][3] = 0.0f;
            }
        }
    };
    auto store_slice = [&](int buf) {
#pragma unroll
        for (int it = 0; it < 4; ++it) {
            h4 hv = { (_Float16)fa[it].x, (_Float16)fa[it].y,
                      (_Float16)fa[it].z, (_Float16)fa[it].w };
            *reinterpret_cast<u32x2*>(&As[buf][(r4 + it * 32) * 40 + c4]) =
                __builtin_bit_cast(u32x2, hv);
        }
#pragma unroll
        for (int it = 0; it < 2; ++it) {
            h4 hv = { (_Float16)fw[it][0], (_Float16)fw[it][1],
                      (_Float16)fw[it][2], (_Float16)fw[it][3] };
            *reinterpret_cast<u32x2*>(&Bs2[buf][kk * 72 + ng + it * 32]) =
                __builtin_bit_cast(u32x2, hv);
        }
    };

    v8f acc[2][2] = {};

    const int nk = K >> 5;             // K/32 slices
    load_slice(0);
    store_slice(0);

    const int ml = lane & 15;
    const int h8 = (lane >> 4) * 8;

    for (int ki = 0; ki < nk; ++ki) {
        const int cur = ki & 1;
        if (ki + 1 < nk) load_slice((ki + 1) << 5);   // prefetch into regs
        __syncthreads();                              // cur buffer ready / safe to reuse next

        union F { u32x4 q[2]; v16h h; };
        F af[2], bf[2];
#pragma unroll
        for (int mt = 0; mt < 2; ++mt) {
            const _Float16* ap = &As[cur][(wm * 32 + mt * 16 + ml) * 40];
            af[mt].q[0] = *reinterpret_cast<const u32x4*>(ap + h8);
            af[mt].q[1] = *reinterpret_cast<const u32x4*>(ap + 16 + h8);
        }
#pragma unroll
        for (int nt = 0; nt < 2; ++nt) {
            const _Float16* bp = &Bs2[cur][wn * 32 + nt * 16 + ml];
#pragma unroll
            for (int s = 0; s < 8; ++s) {
                bf[nt].h[s]     = bp[(h8 + s) * 72];
                bf[nt].h[s + 8] = bp[(16 + h8 + s) * 72];
            }
        }
#pragma unroll
        for (int mt = 0; mt < 2; ++mt)
#pragma unroll
            for (int nt = 0; nt < 2; ++nt)
                acc[mt][nt] = __builtin_amdgcn_wmma_f32_16x16x32_f16(
                    false, af[mt].h, false, bf[nt].h,
                    (short)0, acc[mt][nt], false, false);

        if (ki + 1 < nk) store_slice(1 - cur);        // fill other buffer
    }

    // epilogue: C layout n = lane%16, m = 8*(lane/16) + vgpr
    const int hh    = lane >> 4;
    const int nl0   = lane & 15;
    const bool full = (n0 + 64 <= N);
#pragma unroll
    for (int mt = 0; mt < 2; ++mt)
#pragma unroll
        for (int nt = 0; nt < 2; ++nt)
#pragma unroll
            for (int vv = 0; vv < 8; ++vv) {
                int m = m0 + wm * 32 + mt * 16 + hh * 8 + vv;
                int n = n0 + wn * 32 + nt * 16 + nl0;
                if (full || n < N) {
                    float val = acc[mt][nt][vv];
                    if (bias) val += bias[n];
                    if (act == 1) val = fmaxf(val, 0.0f);
                    else if (act == 2) val = 1.0f / (1.0f + expf(-val));
                    C[(size_t)m * ldc + n] = val;
                }
            }
}

// ---------------------------------------------------------------------------
// GEMV: out[row*ostride] = sigmoid( dot(A[row,:], w) + bias[0] ), wave per row
// ---------------------------------------------------------------------------
__global__ void gemv_sig(const float* __restrict__ A, const float* __restrict__ w,
                         const float* __restrict__ bias, float* __restrict__ out,
                         int K, int ostride)
{
    int row  = blockIdx.x * (blockDim.x >> 5) + (threadIdx.x >> 5);
    int lane = threadIdx.x & 31;
    const float* a = A + (size_t)row * K;
    float s = 0.0f;
    for (int k = lane; k < K; k += 32) s += a[k] * w[k];
#pragma unroll
    for (int o = 16; o; o >>= 1) s += __shfl_xor(s, o, 32);
    if (lane == 0) out[(size_t)row * ostride] = 1.0f / (1.0f + expf(-(s + bias[0])));
}

// ---------------------------------------------------------------------------
// GRU gate math + rank-1 concat term + biases + reparameterized noise.
// gi/gh are [B, 3H] raw GEMM outputs. wlast points at Wih[0][K] (row stride ldw).
// z updated in place: z = (1-zt)*n + zt*z + STD*eps.  H == H_ (compile-time).
// ---------------------------------------------------------------------------
__global__ void gru_gate(const float* __restrict__ gi, const float* __restrict__ gh,
                         const float* __restrict__ bih, const float* __restrict__ bhh,
                         const float* __restrict__ wlast, int ldw,
                         const float* __restrict__ tcol, int tstride, int gt,
                         const float* __restrict__ eps, float* __restrict__ z)
{
    int idx = blockIdx.x * blockDim.x + threadIdx.x;
    int b = idx >> 8;                  // H_ == 256
    int j = idx & (H_ - 1);
    float tv  = tcol[(size_t)b * tstride];
    float tmp = gt ? (tv > 0.5f ? 1.0f : 0.0f) : (tv >= 0.5f ? 1.0f : 0.0f);

    size_t r3 = (size_t)b * 3 * H_;
    float ir = gi[r3 + j]          + tmp * wlast[(size_t)j * ldw]            + bih[j];
    float iz = gi[r3 + H_ + j]     + tmp * wlast[(size_t)(H_ + j) * ldw]     + bih[H_ + j];
    float in_= gi[r3 + 2 * H_ + j] + tmp * wlast[(size_t)(2 * H_ + j) * ldw] + bih[2 * H_ + j];
    float hr = gh[r3 + j]          + bhh[j];
    float hz = gh[r3 + H_ + j]     + bhh[H_ + j];
    float hn = gh[r3 + 2 * H_ + j] + bhh[2 * H_ + j];

    float r  = 1.0f / (1.0f + expf(-(ir + hr)));
    float zt = 1.0f / (1.0f + expf(-(iz + hz)));
    float n  = tanhf(in_ + r * hn);

    size_t o = (size_t)b * H_ + j;
    float hprev = z[o];
    z[o] = (1.0f - zt) * n + zt * hprev + STD_F * eps[o];
}

// latent sampling: lv = softplus(lvlin); latent = mu + exp(lv/2)*eps
__global__ void sample_k(const float* __restrict__ mu, const float* __restrict__ lvlin,
                         const float* __restrict__ eps,
                         float* __restrict__ lv_out, float* __restrict__ latent, int n)
{
    int i = blockIdx.x * blockDim.x + threadIdx.x;
    if (i >= n) return;
    float v  = lvlin[i];
    float lv = fmaxf(v, 0.0f) + log1pf(expf(-fabsf(v)));   // stable softplus
    lv_out[i] = lv;
    latent[i] = mu[i] + expf(0.5f * lv) * eps[i];
}

// ---------------------------------------------------------------------------
// Per-row loss: one wave per batch row (lane doubles as worker index, W_==32)
// rowloss[b] = (log_q - log_p_z) + sum(bce) - log_p_x + log_kl
// ---------------------------------------------------------------------------
__launch_bounds__(256)
__global__ void rowloss_k(const float* __restrict__ rlin, const float* __restrict__ x,
                          const float* __restrict__ preds, const float* __restrict__ workers,
                          const float* __restrict__ t, const float* __restrict__ mu,
                          const float* __restrict__ lv, const float* __restrict__ lat,
                          float* __restrict__ rowloss)
{
    int b    = blockIdx.x * 8 + (threadIdx.x >> 5);
    int lane = threadIdx.x & 31;

    // log_p_x = sum_i x * log(sigmoid(rlin))
    float lpx = 0.0f;
    const float* rl = rlin + (size_t)b * X_;
    const float* xr = x    + (size_t)b * X_;
    for (int i = lane; i < X_; i += 32) {
        float v  = rl[i];
        float ls = fminf(v, 0.0f) - log1pf(expf(-fabsf(v)));   // log sigmoid
        lpx += xr[i] * ls;
    }
    // log_q - log_p_z (constants cancel)
    float lqz = 0.0f;
    for (int i = lane; i < Z_; i += 32) {
        size_t o = (size_t)b * Z_ + i;
        float l = lv[o], m = mu[o], la = lat[o];
        float d = la - m;
        lqz += -0.5f * l - d * d / (2.0f * expf(l)) + 0.5f * la * la;
    }
    // BCE over (w,l) with lane = worker; KL needs per-l mean over workers
    float bce = 0.0f, kl = 0.0f;
    for (int l = 0; l < L_; ++l) {
        float wp = preds[(size_t)l * B_ * W_ + (size_t)b * W_ + lane];
        float wk = workers[(size_t)lane * B_ * L_ + (size_t)b * L_ + l];
        float lp = fmaxf(logf(wp), -100.0f);
        float l1 = fmaxf(log1pf(-wp), -100.0f);
        bce += -(wk * lp + (1.0f - wk) * l1);
        float ws = wk;
#pragma unroll
        for (int o = 16; o; o >>= 1) ws += __shfl_xor(ws, o, 32);
        float wbar = ws * (1.0f / W_);
        float tv   = t[(size_t)b * L_ + l];
        kl += tv * (logf(tv + 1e-6f) - logf(wbar + 1e-6f));    // same on all lanes
    }
#pragma unroll
    for (int o = 16; o; o >>= 1) {
        lpx += __shfl_xor(lpx, o, 32);
        lqz += __shfl_xor(lqz, o, 32);
        bce += __shfl_xor(bce, o, 32);
    }
    if (lane == 0) rowloss[b] = lqz + bce - lpx + kl;
}

__global__ void reduce_loss(const float* __restrict__ rowloss, float* __restrict__ out)
{
    __shared__ float sm[256];
    float s = 0.0f;
    for (int i = threadIdx.x; i < B_; i += 256) s += rowloss[i];
    sm[threadIdx.x] = s;
    __syncthreads();
    for (int st = 128; st; st >>= 1) {
        if (threadIdx.x < st) sm[threadIdx.x] += sm[threadIdx.x + st];
        __syncthreads();
    }
    if (threadIdx.x == 0) out[0] = sm[0] * (1.0f / B_);
}

// ---------------------------------------------------------------------------
extern "C" void kernel_launch(void* const* d_in, const int* in_sizes, int n_in,
                              void* d_out, int out_size, void* d_ws, size_t ws_size,
                              hipStream_t stream)
{
    (void)in_sizes; (void)n_in; (void)out_size; (void)ws_size;
    const float* x       = (const float*)d_in[0];
    const float* workers = (const float*)d_in[1];
    const float* Wf1     = (const float*)d_in[2];
    const float* bf1     = (const float*)d_in[3];
    const float* Wf2     = (const float*)d_in[4];
    const float* bf2     = (const float*)d_in[5];
    const float* eWih    = (const float*)d_in[6];
    const float* eWhh    = (const float*)d_in[7];
    const float* ebih    = (const float*)d_in[8];
    const float* ebhh    = (const float*)d_in[9];
    const float* eWd     = (const float*)d_in[10];
    const float* ebd     = (const float*)d_in[11];
    const float* eWo     = (const float*)d_in[12];
    const float* ebo     = (const float*)d_in[13];
    const float* Wmu     = (const float*)d_in[14];
    const float* bmu     = (const float*)d_in[15];
    const float* Wlv     = (const float*)d_in[16];
    const float* blv     = (const float*)d_in[17];
    const float* dWih    = (const float*)d_in[18];
    const float* dWhh    = (const float*)d_in[19];
    const float* dbih    = (const float*)d_in[20];
    const float* dbhh    = (const float*)d_in[21];
    const float* dWd     = (const float*)d_in[22];
    const float* dbd     = (const float*)d_in[23];
    const float* dWo     = (const float*)d_in[24];
    const float* dbo     = (const float*)d_in[25];
    const float* Wr      = (const float*)d_in[26];
    const float* br      = (const float*)d_in[27];
    const float* enc_eps = (const float*)d_in[28];
    const float* samp_eps= (const float*)d_in[29];
    const float* dec_eps = (const float*)d_in[30];

    float* t_out    = (float*)d_out;            // [B, L]
    float* loss_out = t_out + (size_t)B_ * L_;  // scalar

    float* w = (float*)d_ws;
    float* gi      = w; w += (size_t)B_ * 3 * H_;
    float* gh      = w; w += (size_t)B_ * 3 * H_;
    float* z       = w; w += (size_t)B_ * H_;   // encoder state, reused for decoder
    float* ph      = w; w += (size_t)B_ * H_;   // predict hidden / lvlin temp
    float* muz     = w; w += (size_t)B_ * Z_;
    float* lv      = w; w += (size_t)B_ * Z_;
    float* lat     = w; w += (size_t)B_ * Z_;
    float* preds   = w; w += (size_t)L_ * B_ * W_;
    float* rlin    = w; w += (size_t)B_ * X_;
    float* rowloss = w; w += B_;

    const dim3 blk(256);
    const int H3 = 3 * H_;
    auto gg = [](int N) { return dim3(B_ / 128, (N + 63) / 64); };

    // firstcell: t[:,0] = sigmoid(relu(x@Wf1.T + bf1) @ Wf2.T + bf2)
    gemm_wmma<<<gg(H_), blk, 0, stream>>>(x, X_, Wf1, X_, bf1, ph, H_, B_, H_, X_, 1);
    gemv_sig<<<B_ / 8, 256, 0, stream>>>(ph, Wf2, bf2, t_out, H_, L_);

    hipMemsetAsync(z, 0, (size_t)B_ * H_ * sizeof(float), stream);

    // ---------------- encoder ----------------
    for (int i = 0; i < L_; ++i) {
        const float* Wih = eWih + (size_t)i * H3 * (X_ + 1);
        const float* Whh = eWhh + (size_t)i * H3 * H_;
        gemm_wmma<<<gg(H3), blk, 0, stream>>>(x, X_, Wih, X_ + 1, nullptr, gi, H3, B_, H3, X_, 0);
        gemm_wmma<<<gg(H3), blk, 0, stream>>>(z, H_, Whh, H_,     nullptr, gh, H3, B_, H3, H_, 0);
        gru_gate<<<(B_ * H_) / 256, 256, 0, stream>>>(
            gi, gh, ebih + (size_t)i * H3, ebhh + (size_t)i * H3,
            Wih + X_, X_ + 1, t_out + i, L_, /*gt=*/0,
            enc_eps + (size_t)i * B_ * H_, z);
        if (i < L_ - 1) {  // last step's predict is discarded by the scan
            gemm_wmma<<<gg(H_), blk, 0, stream>>>(
                z, H_, eWd + (size_t)i * H_ * H_, H_, ebd + (size_t)i * H_,
                ph, H_, B_, H_, H_, 1);
            gemv_sig<<<B_ / 8, 256, 0, stream>>>(
                ph, eWo + (size_t)i * H_, ebo + i, t_out + (i + 1), H_, L_);
        }
    }

    // ---------------- latent ----------------
    gemm_wmma<<<gg(Z_), blk, 0, stream>>>(z, H_, Wmu, H_, bmu, muz, Z_, B_, Z_, H_, 0);
    gemm_wmma<<<gg(Z_), blk, 0, stream>>>(z, H_, Wlv, H_, blv, ph,  Z_, B_, Z_, H_, 0);
    sample_k<<<(B_ * Z_) / 256, 256, 0, stream>>>(muz, ph, samp_eps, lv, lat, B_ * Z_);

    hipMemsetAsync(z, 0, (size_t)B_ * H_ * sizeof(float), stream);

    // ---------------- decoder ----------------
    for (int i = 0; i < L_; ++i) {
        const float* Wih = dWih + (size_t)i * H3 * (Z_ + 1);
        const float* Whh = dWhh + (size_t)i * H3 * H_;
        gemm_wmma<<<gg(H3), blk, 0, stream>>>(lat, Z_, Wih, Z_ + 1, nullptr, gi, H3, B_, H3, Z_, 0);
        gemm_wmma<<<gg(H3), blk, 0, stream>>>(z,   H_, Whh, H_,     nullptr, gh, H3, B_, H3, H_, 0);
        gru_gate<<<(B_ * H_) / 256, 256, 0, stream>>>(
            gi, gh, dbih + (size_t)i * H3, dbhh + (size_t)i * H3,
            Wih + Z_, Z_ + 1, t_out + i, L_, /*gt=*/1,
            dec_eps + (size_t)i * B_ * H_, z);
        gemm_wmma<<<gg(H_), blk, 0, stream>>>(
            z, H_, dWd + (size_t)i * H_ * H_, H_, dbd + (size_t)i * H_,
            ph, H_, B_, H_, H_, 1);
        gemm_wmma<<<gg(W_), blk, 0, stream>>>(
            ph, H_, dWo + (size_t)i * W_ * H_, H_, dbo + (size_t)i * W_,
            preds + (size_t)i * B_ * W_, W_, B_, W_, H_, 2);
    }

    // recon linear + fused loss
    gemm_wmma<<<gg(X_), blk, 0, stream>>>(z, H_, Wr, H_, br, rlin, X_, B_, X_, H_, 0);
    rowloss_k<<<B_ / 8, 256, 0, stream>>>(rlin, x, preds, workers, t_out, muz, lv, lat, rowloss);
    reduce_loss<<<1, 256, 0, stream>>>(rowloss, loss_out);
}
